// AtteMtionBlock_53747220742536
// MI455X (gfx1250) — matrix operand order repaired and checked by reading, not compiled, not run on hardware
//
#include <hip/hip_runtime.h>
#include <hip/hip_bf16.h>
#include <cmath>

// ---------------------------------------------------------------------------
// Hierarchical graph-transformer block for MI455X (gfx1250).
// All dense D=128 linears run on the WMMA tensor path:
//   v_wmma_f32_16x16x32_bf16  (bf16 A/B, f32 accumulate)
// W tile is staged in LDS in *WMMA fragment order*, so every B fragment is a
// single aligned 32-byte LDS vector load (ds_load_b128 x2) per lane; B
// fragments are explicitly double-buffered so DS latency overlaps WMMA issue.
// Edge softmax uses f32 global atomics (memory-bound part of the op).
//
// Input layout assumption: setup_inputs() dict insertion order, nested dicts
// flattened recursively in insertion order:
//   0:h  1:graph_h
//   2..11  att_upward   (q.w,q.b,k.w,k.b,v.w,v.b,s.w,s.b,proj.w,proj.b)
//   12..21 att_graph    22..31 att_downward   32..41 att_node
//   42,43 linear_node_1 (w,b)
//   44,45 norm_node_1 (g,b)  46,47 norm_node_2  48,49 ln_1_graph  50,51 ln_2_graph
//   52..55 mlp (l1.w,l1.b,l2.w,l2.b)
//   56:edge_index[2,E] 57:context_edge_index[2,EC] 58:upward[2,N] 59:downward[2,N]
// Output: h5 (N*D f32) || g5 (G*D f32).
// Workspace: ~160 MB (6 N*D + 6 G*D f32 buffers + E + 2N scratch floats).
// ---------------------------------------------------------------------------

typedef __attribute__((ext_vector_type(16))) __bf16 v16bf;
typedef __attribute__((ext_vector_type(8)))  float  v8f;

namespace {
constexpr int Nn  = 50000;
constexpr int Gg  = 1000;
constexpr int Ee  = 500000;
constexpr int ECc = 16000;
constexpr int Dd  = 128;
}

// ---------------------------------------------------------------------------
// GEMM: Y[M x 128] = X[M x 128] @ W[128 x 128] + bias, via WMMA bf16.
// Block = 256 threads (8 waves). Each block: 128 rows x 128 cols.
// Wave w -> rows [blk*128 + 16w, +16). 8 v8f accumulators (one per col tile).
//
// LDS holds W (bf16, 32 KB) pre-shuffled into fragment order:
//   element (K = kc*32 + kr, col n = t*16 + r), kr = half*16 + i, lane = half*16+r
//   -> Wl[ ((kc*8 + t)*32 + lane)*16 + i ]
// so lane's B fragment for fragment index f = kc*8+t is v16bf at
// Wl + (f*32 + lane)*16. Fragments are consumed f = 0..31 with one-deep
// software pipelining (distinct rotating registers -> wait at dscnt<=2).
// ---------------------------------------------------------------------------
__global__ __launch_bounds__(256)
void gemm_bias_kernel(const float* __restrict__ X, const float* __restrict__ W,
                      const float* __restrict__ bias, float* __restrict__ Y,
                      int M) {
  __shared__ __bf16 Wl[Dd * Dd];

  const int tid = threadIdx.x;
  // Stage W -> LDS in fragment order (coalesced global reads).
  for (int g = tid; g < Dd * Dd; g += 256) {
    const int k = g >> 7, n = g & 127;
    const int kc = k >> 5, kr = k & 31;
    const int half = kr >> 4, i = kr & 15;
    const int t = n >> 4, r = n & 15;
    Wl[(((kc * 8 + t) * 32) + half * 16 + r) * 16 + i] = (__bf16)W[g];
  }
  __syncthreads();

  const int wave = tid >> 5;
  const int lane = tid & 31;
  const int r    = lane & 15;   // M (for A) / N (for B,C) within tile
  const int half = lane >> 4;   // K-half selector
  const int row0 = blockIdx.x * 128 + wave * 16;

  v8f acc[8] = {};

  const int arow = (row0 + r < M) ? (row0 + r) : (M - 1);
  const float4* __restrict__ xv =
      (const float4*)(X + (size_t)arow * Dd);
  const v16bf* __restrict__ wfrag = (const v16bf*)Wl;

  // Prime the B-fragment pipeline (fragment index f = kc*8 + t, f = 0..31).
  v16bf bcur = wfrag[lane];

  #pragma unroll
  for (int kc = 0; kc < 4; ++kc) {           // K chunks of 32
    // A fragment, contiguous K order (matches B staging):
    //   elems 0..15 -> K = kbase + 16*half + i
    const int f0 = kc * 8 + 4 * half;
    const float4 p0 = xv[f0];
    const float4 p1 = xv[f0 + 1];
    const float4 p2 = xv[f0 + 2];
    const float4 p3 = xv[f0 + 3];
    v16bf a;
    a[0]  = (__bf16)p0.x;  a[1]  = (__bf16)p0.y;
    a[2]  = (__bf16)p0.z;  a[3]  = (__bf16)p0.w;
    a[4]  = (__bf16)p1.x;  a[5]  = (__bf16)p1.y;
    a[6]  = (__bf16)p1.z;  a[7]  = (__bf16)p1.w;
    a[8]  = (__bf16)p2.x;  a[9]  = (__bf16)p2.y;
    a[10] = (__bf16)p2.z;  a[11] = (__bf16)p2.w;
    a[12] = (__bf16)p3.x;  a[13] = (__bf16)p3.y;
    a[14] = (__bf16)p3.z;  a[15] = (__bf16)p3.w;

    #pragma unroll
    for (int t = 0; t < 8; ++t) {            // 8 column tiles of 16
      const int f = kc * 8 + t;
      // Prefetch next B fragment while WMMA on the current one executes.
      const v16bf bnext = (f < 31) ? wfrag[(f + 1) * 32 + lane] : bcur;
      acc[t] = __builtin_amdgcn_wmma_f32_16x16x32_bf16(
          false, a, false, bcur, (short)0, acc[t], false, false);
      bcur = bnext;
    }
  }

  // C/D layout: lane(half) elem j -> row = 8*half + j, col = t*16 + r
  #pragma unroll
  for (int t = 0; t < 8; ++t) {
    const int col = t * 16 + r;
    const float bv = bias[col];
    #pragma unroll
    for (int j = 0; j < 8; ++j) {
      const int row = row0 + half * 8 + j;
      if (row < M) Y[(size_t)row * Dd + col] = acc[t][j] + bv;
    }
  }
}

// ---------------------------------------------------------------------------
// Edge softmax helpers
// ---------------------------------------------------------------------------
__device__ inline void atomicMaxF(float* addr, float val) {
  if (val >= 0.0f) atomicMax((int*)addr, __float_as_int(val));
  else             atomicMin((unsigned int*)addr, __float_as_uint(val));
}

__global__ __launch_bounds__(256)
void fill_kernel(float* __restrict__ p, float val, int n) {
  const int i = blockIdx.x * 256 + threadIdx.x;
  if (i < n) p[i] = val;
}

// Pass 1: logits + segment max (warp per edge)
__global__ __launch_bounds__(256)
void edge_logits_max_kernel(const float* __restrict__ q, const float* __restrict__ k,
                            const int* __restrict__ es, const int* __restrict__ ed,
                            float* __restrict__ lbuf, float* __restrict__ mbuf, int ne) {
  const int e    = blockIdx.x * 8 + (threadIdx.x >> 5);
  const int lane = threadIdx.x & 31;
  if (e >= ne) return;
  const int s = es[e], d = ed[e];
  const float* qr = q + (size_t)d * Dd;
  const float* kr = k + (size_t)s * Dd;
  float sum = 0.0f;
  #pragma unroll
  for (int i = 0; i < 4; ++i) sum += qr[lane + 32 * i] * kr[lane + 32 * i];
  #pragma unroll
  for (int o = 16; o; o >>= 1) sum += __shfl_down(sum, o, 32);
  if (lane == 0) {
    const float l = sum * 0.08838834765f;   // 1/sqrt(128)
    lbuf[e] = l;
    atomicMaxF(mbuf + d, l);
  }
}

// Pass 2: ex = exp(l - m[ed]) (m fixed to 0 if -inf), denom += ex
__global__ __launch_bounds__(256)
void edge_exp_sum_kernel(float* __restrict__ lbuf, const float* __restrict__ mbuf,
                         const int* __restrict__ ed, float* __restrict__ dbuf, int ne) {
  const int e = blockIdx.x * 256 + threadIdx.x;
  if (e >= ne) return;
  const int d = ed[e];
  float m = mbuf[d];
  if (!(m > -3.0e38f)) m = 0.0f;            // isfinite(-inf) -> 0 fixup
  const float ex = __expf(lbuf[e] - m);
  lbuf[e] = ex;
  atomicAdd(dbuf + d, ex);
}

// Pass 3: agg[ed] += v[es] * alpha (warp per edge); agg pre-init = lin(x_dst,s)
__global__ __launch_bounds__(256)
void edge_scatter_kernel(const float* __restrict__ v, const float* __restrict__ exb,
                         const float* __restrict__ dbuf,
                         const int* __restrict__ es, const int* __restrict__ ed,
                         float* __restrict__ agg, int ne) {
  const int e    = blockIdx.x * 8 + (threadIdx.x >> 5);
  const int lane = threadIdx.x & 31;
  if (e >= ne) return;
  const int s = es[e], d = ed[e];
  const float alpha = exb[e] / fmaxf(dbuf[d], 1e-16f);
  const float* vr = v + (size_t)s * Dd;
  float* ar = agg + (size_t)d * Dd;
  #pragma unroll
  for (int i = 0; i < 4; ++i) {
    const int c = lane + 32 * i;
    atomicAdd(ar + c, vr[c] * alpha);
  }
}

// ---------------------------------------------------------------------------
// Fused residual + LayerNorm, warp per row
// ---------------------------------------------------------------------------
__global__ __launch_bounds__(256)
void ln_residual_kernel(const float* __restrict__ a, const float* __restrict__ res,
                        const float* __restrict__ gam, const float* __restrict__ bet,
                        float* __restrict__ y, int M) {
  const int row  = blockIdx.x * 8 + (threadIdx.x >> 5);
  const int lane = threadIdx.x & 31;
  if (row >= M) return;
  const float* ar = a   + (size_t)row * Dd;
  const float* rr = res + (size_t)row * Dd;
  float x[4];
  float mean = 0.0f;
  #pragma unroll
  for (int i = 0; i < 4; ++i) { x[i] = ar[lane + 32 * i] + rr[lane + 32 * i]; mean += x[i]; }
  #pragma unroll
  for (int o = 16; o; o >>= 1) mean += __shfl_down(mean, o, 32);
  mean = __shfl(mean, 0, 32) * (1.0f / 128.0f);
  float var = 0.0f;
  #pragma unroll
  for (int i = 0; i < 4; ++i) { x[i] -= mean; var += x[i] * x[i]; }
  #pragma unroll
  for (int o = 16; o; o >>= 1) var += __shfl_down(var, o, 32);
  var = __shfl(var, 0, 32) * (1.0f / 128.0f);
  const float inv = rsqrtf(var + 1e-5f);
  float* yr = y + (size_t)row * Dd;
  #pragma unroll
  for (int i = 0; i < 4; ++i) {
    const int c = lane + 32 * i;
    yr[c] = x[i] * inv * gam[c] + bet[c];
  }
}

// NewGELU (tanh approximation), in place
__global__ __launch_bounds__(256)
void gelu_kernel(float* __restrict__ x, int n) {
  const int i = blockIdx.x * 256 + threadIdx.x;
  if (i >= n) return;
  const float v = x[i];
  const float t = tanhf(0.7978845608f * (v + 0.044715f * v * v * v));
  x[i] = 0.5f * v * (1.0f + t);
}

// ---------------------------------------------------------------------------
// Host-side orchestration
// ---------------------------------------------------------------------------
static inline int cdiv(int a, int b) { return (a + b - 1) / b; }

static void run_gemm(const float* X, const void* W, const void* b, float* Y,
                     int M, hipStream_t s) {
  gemm_bias_kernel<<<cdiv(M, 128), 256, 0, s>>>(
      X, (const float*)W, (const float*)b, Y, M);
}

// One full TransformerConv + proj:  out = lin(tconv(xsrc,xdst,es,ed), proj)
static void run_att(const float* xsrc, int Ns, const float* xdst, int Nd,
                    const int* es, const int* ed, int ne,
                    void* const* P, int pb,
                    float* q, float* k, float* v, float* agg, float* out,
                    float* exb, float* mb, float* db, hipStream_t s) {
  run_gemm(xdst, P[pb + 0], P[pb + 1], q,   Nd, s);   // q = lin(x_dst)
  run_gemm(xsrc, P[pb + 2], P[pb + 3], k,   Ns, s);   // k = lin(x_src)
  run_gemm(xsrc, P[pb + 4], P[pb + 5], v,   Ns, s);   // v = lin(x_src)
  run_gemm(xdst, P[pb + 6], P[pb + 7], agg, Nd, s);   // agg = lin(x_dst, s) (root weight)
  fill_kernel<<<cdiv(Nd, 256), 256, 0, s>>>(mb, -__builtin_huge_valf(), Nd);
  fill_kernel<<<cdiv(Nd, 256), 256, 0, s>>>(db, 0.0f, Nd);
  edge_logits_max_kernel<<<cdiv(ne, 8), 256, 0, s>>>(q, k, es, ed, exb, mb, ne);
  edge_exp_sum_kernel<<<cdiv(ne, 256), 256, 0, s>>>(exb, mb, ed, db, ne);
  edge_scatter_kernel<<<cdiv(ne, 8), 256, 0, s>>>(v, exb, db, es, ed, agg, ne);
  run_gemm(agg, P[pb + 8], P[pb + 9], out, Nd, s);    // out = lin(agg, proj)
}

extern "C" void kernel_launch(void* const* d_in, const int* in_sizes, int n_in,
                              void* d_out, int out_size, void* d_ws, size_t ws_size,
                              hipStream_t stream) {
  (void)in_sizes; (void)n_in; (void)out_size; (void)ws_size;

  const float* h  = (const float*)d_in[0];
  const float* gh = (const float*)d_in[1];
  const int* ei  = (const int*)d_in[56];
  const int* cei = (const int*)d_in[57];
  const int* up  = (const int*)d_in[58];
  const int* dn  = (const int*)d_in[59];

  const size_t ND = (size_t)Nn * Dd, GD = (size_t)Gg * Dd;
  float* base = (float*)d_ws;
  float* nb[6];
  for (int i = 0; i < 6; ++i) nb[i] = base + (size_t)i * ND;
  float* gB = base + 6 * ND;
  float* gb[6];
  for (int i = 0; i < 6; ++i) gb[i] = gB + (size_t)i * GD;
  float* exb = gB + 6 * GD;   // E floats (per-edge exp scratch)
  float* mb  = exb + Ee;      // N floats (segment max; G fits in N)
  float* db  = mb + Nn;       // N floats (segment denom)

  float* out_h = (float*)d_out;        // h5 : N x D
  float* out_g = out_h + ND;           // g5 : G x D

  // --- Stage A: g1 = att_upward(h -> graph_h), edges = upward (ne = N)
  run_att(h, Nn, gh, Gg, up, up + Nn, Nn, d_in, 2,
          /*q*/gb[0], /*k*/nb[1], /*v*/nb[2], /*agg*/gb[1], /*g1*/gb[2],
          exb, mb, db, stream);

  // --- Stage B: h_lin = lin(h, linear_node_1)
  run_gemm(h, d_in[42], d_in[43], nb[3], Nn, stream);

  // --- Stage C: g2 = att_graph(g1 -> g1), edges = context (ne = EC)
  run_att(gb[2], Gg, gb[2], Gg, cei, cei + ECc, ECc, d_in, 12,
          gb[0], gb[1], gb[3], gb[4], /*g2*/gb[5], exb, mb, db, stream);
  // g3 = LN(g2 + graph_h)
  ln_residual_kernel<<<cdiv(Gg, 8), 256, 0, stream>>>(
      gb[5], gh, (const float*)d_in[48], (const float*)d_in[49], gb[1], Gg);

  // --- Stage D: h2 = att_downward(g3 -> h_lin), edges = downward (ne = N)
  run_att(gb[1], Gg, nb[3], Nn, dn, dn + Nn, Nn, d_in, 22,
          /*q*/nb[1], /*k*/gb[0], /*v*/gb[3], /*agg*/nb[2], /*h2*/nb[4],
          exb, mb, db, stream);
  // h3 = LN(h2 + h)
  ln_residual_kernel<<<cdiv(Nn, 8), 256, 0, stream>>>(
      nb[4], h, (const float*)d_in[44], (const float*)d_in[45], nb[0], Nn);

  // --- Stage E: h4 = att_node(h3 -> h3), edges = edge_index (ne = E)
  run_att(nb[0], Nn, nb[0], Nn, ei, ei + Ee, Ee, d_in, 32,
          nb[1], nb[2], nb[3], nb[4], /*h4*/nb[5], exb, mb, db, stream);
  // h5 = LN(h4 + h3) -> out
  ln_residual_kernel<<<cdiv(Nn, 8), 256, 0, stream>>>(
      nb[5], nb[0], (const float*)d_in[46], (const float*)d_in[47], out_h, Nn);

  // --- Stage F: graph MLP on g3
  run_gemm(gb[1], d_in[52], d_in[53], gb[0], Gg, stream);
  gelu_kernel<<<cdiv((int)GD, 256), 256, 0, stream>>>(gb[0], (int)GD);
  run_gemm(gb[0], d_in[54], d_in[55], gb[3], Gg, stream);
  // g5 = LN(g4 + g3) -> out
  ln_residual_kernel<<<cdiv(Gg, 8), 256, 0, stream>>>(
      gb[3], gb[1], (const float*)d_in[50], (const float*)d_in[51], out_g, Gg);
}
